// CRF_8916352106783
// MI455X (gfx1250) — matrix operand order repaired and checked by reading, not compiled, not run on hardware
//
#include <hip/hip_runtime.h>
#include <hip/hip_bf16.h>
#include <math.h>

// CRF NLL for B=256, S=2048, T=48 on MI455X (gfx1250, wave32).
//
// Roofline: emissions (96 MB) read once -> 4.1 us floor @ 23.3 TB/s; compute
// ~5 GFLOP effective. Binding constraint = S=2048 sequential scan steps, so we
// minimize dependent-path work per step and express the T x T logsumexp
// contraction as a max-normalized f16 WMMA matmul with f32 accumulation:
//   alpha'[r,j] = em + M_r + log( (exp(alpha-M_r)) @ exp(trans) )[r,j]
// P in (0,1], E = exp(N(0,1)) in [~e^-5, e^5]: both safely f16.
// E (B-operand) is register-resident per wave for the entire scan; the next
// step's emissions tile is DMA'd into an LDS double buffer with
// GLOBAL_LOAD_ASYNC_TO_LDS_B128 (ASYNCcnt) one iteration ahead.

#define CRF_B 256
#define CRF_S 2048
#define CRF_T 48
#define CRF_TP 64   // K padded to 64 (two 16x16x32 WMMA K-steps)

typedef float    v8f  __attribute__((ext_vector_type(8)));
typedef _Float16 v16h __attribute__((ext_vector_type(16)));
typedef _Float16 v8h  __attribute__((ext_vector_type(8)));

// ---------------------------------------------------------------------------
// Forward scan: 16 blocks (one 16-row batch tile each) x 96 threads (3 waves,
// one 16-col tag tile per wave). Writes fwd[b] (logZ per batch) to fwd_out.
// ---------------------------------------------------------------------------
__global__ __launch_bounds__(96)
void crf_forward_kernel(const float* __restrict__ em,
                        const int*   __restrict__ mask,
                        const float* __restrict__ trans,
                        const float* __restrict__ start,
                        const float* __restrict__ endt,
                        float*       __restrict__ fwd_out)
{
    __shared__ __attribute__((aligned(64)))  float    Acm[CRF_T][16];   // alpha, col-major [tag][row]
    __shared__ __attribute__((aligned(128))) _Float16 P[16][CRF_TP];    // exp(alpha - M), A-operand source
    __shared__ __attribute__((aligned(16)))  float    EmBuf[2][16][CRF_T]; // async emissions double buffer
    __shared__ float PMax[16][6];
    __shared__ float Mrow[16];
    __shared__ int   MaskS[16];

    const int tid  = threadIdx.x;
    const int lane = tid & 31;
    const int wave = tid >> 5;          // tag-column tile: cols 16*wave .. +15
    const int hh   = (lane >> 4) & 1;   // lane half (WMMA layout)
    const int lcol = lane & 15;
    const int bt   = blockIdx.x;        // batch tile
    const int b0   = bt * 16;
    const int col  = lcol + 16 * wave;  // this lane's N (tag col) in C/D layout
    const int row0 = hh * 8;            // D regs v=0..7 hold batch rows row0..+7

    // Async-copy geometry: 16 rows x 192 B = 12 x b128 segments per row,
    // 192 segments over 2 issues x 96 lanes.
    const int seg_r0  = tid / 12,        seg_c0  = tid % 12;         // issue 0
    const int seg_r1  = (tid + 96) / 12, seg_c1  = (tid + 96) % 12;  // issue 1

#define CRF_ISSUE_EM(snext, buf)                                               \
    do {                                                                       \
        const float* g0 = em + ((size_t)(b0 + seg_r0) * CRF_S + (snext)) * CRF_T + seg_c0 * 4; \
        const float* g1 = em + ((size_t)(b0 + seg_r1) * CRF_S + (snext)) * CRF_T + seg_c1 * 4; \
        unsigned l0 = (unsigned)(uintptr_t)&EmBuf[(buf)][seg_r0][seg_c0 * 4];  \
        unsigned l1 = (unsigned)(uintptr_t)&EmBuf[(buf)][seg_r1][seg_c1 * 4];  \
        asm volatile("global_load_async_to_lds_b128 %0, %1, off"               \
                     :: "v"(l0), "v"(g0) : "memory");                          \
        asm volatile("global_load_async_to_lds_b128 %0, %1, off"               \
                     :: "v"(l1), "v"(g1) : "memory");                          \
    } while (0)

    // Kick off the DMA for step s=1 immediately (overlaps all init work).
    CRF_ISSUE_EM(1, 1);

    // --- Build register-resident B operands: E[i][j] = exp(trans[i][j]) -----
    // B 32x16 f16 layout: lanes 0-15 -> N=lane, K = 0..15 across halves;
    //                     lanes 16-31 -> N=lane-16, K = 16..31.
    v16h bop0, bop1;
    for (int k = 0; k < 16; ++k) {
        int i0 = hh * 16 + k;           // K for first WMMA  (0..31)
        int i1 = 32 + hh * 16 + k;      // K for second WMMA (32..63)
        bop0[k] = (_Float16)expf(trans[i0 * CRF_T + col]);
        bop1[k] = (i1 < CRF_T) ? (_Float16)expf(trans[i1 * CRF_T + col])
                               : (_Float16)0.0f;
    }

    // --- Zero the padded K region of P once (never rewritten) ---------------
    for (int idx = tid; idx < 16 * (CRF_TP - CRF_T); idx += 96) {
        int r = idx / (CRF_TP - CRF_T);
        int c = CRF_T + idx % (CRF_TP - CRF_T);
        P[r][c] = (_Float16)0.0f;
    }

    // --- Init alpha (step 0): start_transitions + emissions[:,0] ------------
    v8f alpha;
    for (int v = 0; v < 8; ++v) {
        int r = row0 + v;
        alpha[v] = start[col] + em[((size_t)(b0 + r)) * CRF_S * CRF_T + col];
    }

    const int rr = tid & 15;            // row handled in stages A/B
    const int gg = tid >> 4;            // j-group 0..5
    const int j0 = gg * 8;

    // --- Sequential scan over s = 1 .. S-1 ----------------------------------
    for (int s = 1; s < CRF_S; ++s) {
        // Prefetch step s+1 into the other half of the double buffer (async DMA).
        if (s + 1 < CRF_S) CRF_ISSUE_EM(s + 1, (s + 1) & 1);

        // Stage 0: publish alpha to LDS, col-major (contiguous per lane).
        *(v8f*)(&Acm[col][row0]) = alpha;
        __syncthreads();

        // Stage A: cache this thread's 8 cells, partial row max, mask fetch.
        float av[8];
        float pm = Acm[j0][rr];
        av[0] = pm;
        for (int k = 1; k < 8; ++k) {
            av[k] = Acm[j0 + k][rr];
            pm = fmaxf(pm, av[k]);
        }
        PMax[rr][gg] = pm;
        if (gg == 1) MaskS[rr] = mask[(size_t)(b0 + rr) * CRF_S + s];
        __syncthreads();

        // Stage B: fold 6 partial maxima, P[r][j] = exp(alpha - M_r) as f16.
        float mr = PMax[rr][0];
        for (int q = 1; q < 6; ++q) mr = fmaxf(mr, PMax[rr][q]);
        if (gg == 0) Mrow[rr] = mr;
        {
            v8h p;
            for (int k = 0; k < 8; ++k)
                p[k] = (_Float16)__expf(av[k] - mr);
            *(v8h*)(&P[rr][j0]) = p;
        }
        // The oldest async pair (this step's emissions) must be in LDS before
        // the barrier that releases stage C.
        if (s + 1 < CRF_S) asm volatile("s_wait_asynccnt 0x2" ::: "memory");
        else               asm volatile("s_wait_asynccnt 0x0" ::: "memory");
        __syncthreads();

        // Stage C: A-operand loads (contiguous K order, same as bop0/bop1), 2 WMMAs.
        // lane half 0: K = 0..15 (+32); half 1: K = 16..31 (+32).
        const v8h* prow = (const v8h*)(&P[lcol][0]);
        v8h a0lo = prow[0 + 2 * hh], a0hi = prow[1 + 2 * hh];
        v8h a1lo = prow[4 + 2 * hh], a1hi = prow[5 + 2 * hh];
        v16h A0 = __builtin_shufflevector(a0lo, a0hi,
                    0,1,2,3,4,5,6,7,8,9,10,11,12,13,14,15);
        v16h A1 = __builtin_shufflevector(a1lo, a1hi,
                    0,1,2,3,4,5,6,7,8,9,10,11,12,13,14,15);

        v8f acc = {0.f,0.f,0.f,0.f,0.f,0.f,0.f,0.f};
        acc = __builtin_amdgcn_wmma_f32_16x16x32_f16(
                  false, A0, false, bop0, (short)0, acc, false, false);
        acc = __builtin_amdgcn_wmma_f32_16x16x32_f16(
                  false, A1, false, bop1, (short)0, acc, false, false);

        const float* emrow = &EmBuf[s & 1][row0][col];
        for (int v = 0; v < 8; ++v) {
            int r = row0 + v;
            float e    = emrow[v * CRF_T];          // LDS, conflict-free
            float anew = e + Mrow[r] + __logf(acc[v]);
            alpha[v]   = MaskS[r] ? anew : alpha[v];
        }
    }

    // --- fwd[b] = logsumexp_j(alpha[b,j] + end_transitions[j]) --------------
    *(v8f*)(&Acm[col][row0]) = alpha;
    __syncthreads();
    if (tid < 16) {
        int r = tid;
        float m = Acm[0][r] + endt[0];
        for (int j = 1; j < CRF_T; ++j) m = fmaxf(m, Acm[j][r] + endt[j]);
        float sum = 0.f;
        for (int j = 0; j < CRF_T; ++j) sum += __expf(Acm[j][r] + endt[j] - m);
        fwd_out[b0 + r] = m + __logf(sum);
    }
#undef CRF_ISSUE_EM
}

// ---------------------------------------------------------------------------
// Gold path score: one block per batch element, parallel over s + reduction.
// ---------------------------------------------------------------------------
__global__ __launch_bounds__(256)
void crf_gold_kernel(const float* __restrict__ em,
                     const int*   __restrict__ tags,
                     const int*   __restrict__ mask,
                     const float* __restrict__ trans,
                     const float* __restrict__ start,
                     const float* __restrict__ endt,
                     float*       __restrict__ gold_out)
{
    __shared__ float sred[256];
    __shared__ int   cred[256];
    const int b = blockIdx.x, tid = threadIdx.x;

    float acc = 0.f;
    int   cnt = 0;
    for (int s = tid; s < CRF_S; s += 256) {
        int m = mask[(size_t)b * CRF_S + s];
        cnt += m;
        if (s >= 1) {
            int tp = tags[(size_t)b * CRF_S + s - 1];
            int tc = tags[(size_t)b * CRF_S + s];
            float v = trans[tp * CRF_T + tc]
                    + em[(((size_t)b * CRF_S) + s) * CRF_T + tc];
            acc += m ? v : 0.f;
        }
    }
    sred[tid] = acc; cred[tid] = cnt;
    __syncthreads();
    for (int off = 128; off > 0; off >>= 1) {
        if (tid < off) { sred[tid] += sred[tid + off]; cred[tid] += cred[tid + off]; }
        __syncthreads();
    }
    if (tid == 0) {
        int t0 = tags[(size_t)b * CRF_S];
        float g = start[t0] + em[(size_t)b * CRF_S * CRF_T + t0] + sred[0];
        int last = cred[0] - 1;
        if (last < 0) last = 0;
        g += endt[tags[(size_t)b * CRF_S + last]];
        gold_out[b] = g;
    }
}

// ---------------------------------------------------------------------------
// Final: mean(fwd - gold) over B=256 -> d_out[0]
// ---------------------------------------------------------------------------
__global__ __launch_bounds__(256)
void crf_reduce_kernel(const float* __restrict__ fwd,
                       const float* __restrict__ gold,
                       float*       __restrict__ out)
{
    __shared__ float sred[256];
    const int tid = threadIdx.x;
    sred[tid] = fwd[tid] - gold[tid];
    __syncthreads();
    for (int off = 128; off > 0; off >>= 1) {
        if (tid < off) sred[tid] += sred[tid + off];
        __syncthreads();
    }
    if (tid == 0) out[0] = sred[0] / (float)CRF_B;
}

extern "C" void kernel_launch(void* const* d_in, const int* in_sizes, int n_in,
                              void* d_out, int out_size, void* d_ws, size_t ws_size,
                              hipStream_t stream)
{
    const float* em    = (const float*)d_in[0];   // (B,S,T) f32
    const int*   tags  = (const int*)  d_in[1];   // (B,S) int32
    const int*   mask  = (const int*)  d_in[2];   // (B,S) int32
    const float* trans = (const float*)d_in[3];   // (T,T) f32
    const float* start = (const float*)d_in[4];   // (T,) f32
    const float* endt  = (const float*)d_in[5];   // (T,) f32

    float* fwd  = (float*)d_ws;            // [256]
    float* gold = fwd + CRF_B;             // [256]
    float* out  = (float*)d_out;

    crf_gold_kernel<<<CRF_B, 256, 0, stream>>>(em, tags, mask, trans, start, endt, gold);
    crf_forward_kernel<<<CRF_B / 16, 96, 0, stream>>>(em, mask, trans, start, endt, fwd);
    crf_reduce_kernel<<<1, 256, 0, stream>>>(fwd, gold, out);
}
